// LigandPocketDiffPoolingHead_60438779789615
// MI455X (gfx1250) — matrix operand order repaired and checked by reading, not compiled, not run on hardware
//
#include <hip/hip_runtime.h>
#include <stdint.h>

// Problem constants (match the reference module constants).
#define NN 10000
#define EE 160000
#define HH 256
#define GG 64
#define PER_HEAD (NN + EE)          // 170000 rows per head (nodes then edges)
#define TOTAL_ROWS (3 * PER_HEAD)   // 510000 virtual rows, shared mlp1 weights

typedef __attribute__((ext_vector_type(16))) __bf16 v16bf;
typedef __attribute__((ext_vector_type(8)))  float  v8f;

union Frag { uint4 u[2]; v16bf v; };

__device__ __forceinline__ unsigned short f2bf(float x) {
    uint32_t u = __float_as_uint(x);
    u += 0x7FFFu + ((u >> 16) & 1u);          // round-to-nearest-even
    return (unsigned short)(u >> 16);
}

__device__ __forceinline__ float silu(float x) {
    return x / (1.0f + __expf(-x));
}

// ---------------------------------------------------------------------------
// Kernel 0: pack mlp1 weights (fp32 HxH row-major) into bf16 B-fragment layout
//   Wp[layer][(ks*16 + ntile)*32 + lane][v]  (uint32 = bf16 pair k, k+1)
//   with n = ntile*16 + lane%16,  k = ks*32 + (lane/16)*16 + 2*v
// and zero the pooled accumulator (6*G*H floats == 98304 == 3*32768 threads).
// ---------------------------------------------------------------------------
__global__ __launch_bounds__(256) void pack_weights_zero(
    const float* __restrict__ W1, const float* __restrict__ W2,
    const float* __restrict__ W3, uint32_t* __restrict__ Wp,
    float* __restrict__ pooled)
{
    int idx = blockIdx.x * 256 + threadIdx.x;   // 0 .. 98303
    pooled[idx] = 0.0f;

    int l = idx >> 15;            // layer
    int r = idx & 32767;
    const float* W = (l == 0) ? W1 : ((l == 1) ? W2 : W3);
    int v     = r & 7;
    int lane  = (r >> 3) & 31;
    int ntile = (r >> 8) & 15;
    int ks    = r >> 12;
    int n = ntile * 16 + (lane & 15);
    int k = ks * 32 + (lane >> 4) * 16 + v * 2;
    uint32_t lo = f2bf(W[(size_t)k * HH + n]);
    uint32_t hi = f2bf(W[(size_t)(k + 1) * HH + n]);
    Wp[idx] = lo | (hi << 16);
}

// ---------------------------------------------------------------------------
// One mlp1 layer on a 64x256 LDS tile: out = silu(in @ W + b), bf16 in/out,
// fp32 WMMA accumulation.  8 waves: wm=wave&3 row-tile, wn=wave>>2 col-half.
// ---------------------------------------------------------------------------
__device__ __forceinline__ void mlp_layer_accum(
    const unsigned short* __restrict__ inLds,
    const uint32_t* __restrict__ Wp, v8f acc[8],
    int lane, int wm, int wn)
{
    const int mrow = wm * 16 + (lane & 15);
    const int half = lane >> 4;
#pragma unroll
    for (int ks = 0; ks < 8; ++ks) {
        Frag a;
        const unsigned short* ap = inLds + mrow * 256 + ks * 32 + half * 16;
        a.u[0] = *(const uint4*)(ap);        // K 0..7   of this half
        a.u[1] = *(const uint4*)(ap + 8);    // K 8..15  of this half
#pragma unroll
        for (int c = 0; c < 8; ++c) {
            int ntile = wn * 8 + c;
            const uint32_t* bp = Wp + (((ks * 16 + ntile) * 32 + lane) << 3);
            Frag b;
            b.u[0] = *(const uint4*)(bp);
            b.u[1] = *(const uint4*)(bp + 4);
            acc[c] = __builtin_amdgcn_wmma_f32_16x16x32_bf16(
                false, a.v, false, b.v, (short)0, acc[c], false, false);
        }
    }
}

__device__ __forceinline__ void mlp_layer_to_lds(
    const unsigned short* __restrict__ inLds,
    unsigned short* __restrict__ outLds,
    const uint32_t* __restrict__ Wp, const float* __restrict__ bias,
    int lane, int wm, int wn)
{
    v8f acc[8] = {};
    mlp_layer_accum(inLds, Wp, acc, lane, wm, wn);
    const int half = lane >> 4;
#pragma unroll
    for (int c = 0; c < 8; ++c) {
        int n = (wn * 8 + c) * 16 + (lane & 15);
        float bb = bias[n];
#pragma unroll
        for (int v = 0; v < 8; ++v) {
            int m = wm * 16 + v + 8 * half;
            outLds[m * 256 + n] = f2bf(silu(acc[c][v] + bb));
        }
    }
}

__device__ __forceinline__ void mlp_layer_scatter(
    const unsigned short* __restrict__ inLds,
    const int* __restrict__ destL,
    const uint32_t* __restrict__ Wp, const float* __restrict__ bias,
    float* __restrict__ pooled,
    int lane, int wm, int wn)
{
    v8f acc[8] = {};
    mlp_layer_accum(inLds, Wp, acc, lane, wm, wn);
    const int half = lane >> 4;
#pragma unroll
    for (int c = 0; c < 8; ++c) {
        int n = (wn * 8 + c) * 16 + (lane & 15);
        float bb = bias[n];
#pragma unroll
        for (int v = 0; v < 8; ++v) {
            int m = wm * 16 + v + 8 * half;
            int d = destL[m];
            float y = silu(acc[c][v] + bb);
            if (d >= 0) atomicAdd(pooled + d + n, y);   // L2-resident
        }
    }
}

// ---------------------------------------------------------------------------
// Main fused kernel: 1 block = 64 virtual rows through all 3 mlp1 layers,
// then atomic scatter into pooled[6][G][H].
// ---------------------------------------------------------------------------
__global__ __launch_bounds__(256) void fused_mlp1_scatter(
    const float* __restrict__ nf0, const float* __restrict__ ef0,
    const int* __restrict__ ei0, const int* __restrict__ bt0,
    const float* __restrict__ nf1, const float* __restrict__ ef1,
    const int* __restrict__ ei1, const int* __restrict__ bt1,
    const float* __restrict__ nf2, const float* __restrict__ ef2,
    const int* __restrict__ ei2, const int* __restrict__ bt2,
    const uint32_t* __restrict__ Wp,
    const float* __restrict__ bia1, const float* __restrict__ bia2,
    const float* __restrict__ bia3,
    float* __restrict__ pooled)
{
    __shared__ unsigned short actA[64 * 256];   // 32 KB bf16 tile
    __shared__ unsigned short actB[64 * 256];   // 32 KB bf16 tile
    __shared__ int destL[64];

    const int tid = threadIdx.x;
    const long base = (long)blockIdx.x * 64;

    // Stage 1: gather 64 rows (fp32) -> bf16 LDS, fully coalesced float4 loads.
#pragma unroll
    for (int j = 0; j < 16; ++j) {
        int lin4 = j * 256 + tid;       // 4096 float4 slots in the tile
        int row  = lin4 >> 6;           // 64 float4 per row
        int col4 = lin4 & 63;
        long vr = base + row;
        float4 val = make_float4(0.f, 0.f, 0.f, 0.f);
        if (vr < TOTAL_ROWS) {
            int head = (int)(vr / PER_HEAD);
            int rem  = (int)(vr - (long)head * PER_HEAD);
            const float* nf = (head == 0) ? nf0 : ((head == 1) ? nf1 : nf2);
            const float* ef = (head == 0) ? ef0 : ((head == 1) ? ef1 : ef2);
            const float* src = (rem < NN) ? (nf + (size_t)rem * HH)
                                          : (ef + (size_t)(rem - NN) * HH);
            val = ((const float4*)src)[col4];
        }
        uint2 p;
        p.x = (uint32_t)f2bf(val.x) | ((uint32_t)f2bf(val.y) << 16);
        p.y = (uint32_t)f2bf(val.z) | ((uint32_t)f2bf(val.w) << 16);
        *(uint2*)&actA[row * 256 + col4 * 4] = p;
    }
    // Per-row scatter destination: nodes -> batch[i]; edges -> batch[recv[e]].
    if (tid < 64) {
        long vr = base + tid;
        int d = -1;
        if (vr < TOTAL_ROWS) {
            int head = (int)(vr / PER_HEAD);
            int rem  = (int)(vr - (long)head * PER_HEAD);
            const int* ei = (head == 0) ? ei0 : ((head == 1) ? ei1 : ei2);
            const int* bt = (head == 0) ? bt0 : ((head == 1) ? bt1 : bt2);
            int g, ie;
            if (rem < NN) { g = bt[rem]; ie = 0; }
            else          { int recv = ei[EE + (rem - NN)]; g = bt[recv]; ie = 1; }
            d = ((head * 2 + ie) * GG + g) * HH;
        }
        destL[tid] = d;
    }
    __syncthreads();

    const int lane = tid & 31;
    const int w = tid >> 5;
    const int wm = w & 3;      // row tile (16 rows)
    const int wn = w >> 2;     // column half (128 cols)

    mlp_layer_to_lds(actA, actB, Wp,          bia1, lane, wm, wn);
    __syncthreads();
    mlp_layer_to_lds(actB, actA, Wp + 32768,  bia2, lane, wm, wn);
    __syncthreads();
    mlp_layer_scatter(actA, destL, Wp + 65536, bia3, pooled, lane, wm, wn);
}

// ---------------------------------------------------------------------------
// mlp2 on the tiny pooled matrix (3 heads x 64 graphs) — exact fp32.
// ---------------------------------------------------------------------------
__global__ __launch_bounds__(256) void head_mlp2_l1(
    const float* __restrict__ pooled, const float* __restrict__ V1,
    const float* __restrict__ c1, float* __restrict__ h1)
{
    __shared__ float x[512];
    int head = blockIdx.x >> 6;
    int g    = blockIdx.x & 63;
    int n    = threadIdx.x;
    x[n]       = pooled[((head * 2 + 0) * GG + g) * HH + n];
    x[n + 256] = pooled[((head * 2 + 1) * GG + g) * HH + n];
    __syncthreads();
    float acc = c1[n];
#pragma unroll 4
    for (int k = 0; k < 512; ++k) acc = fmaf(x[k], V1[(size_t)k * HH + n], acc);
    h1[((size_t)head * GG + g) * HH + n] = silu(acc);
}

__global__ __launch_bounds__(256) void head_mlp2_l2(
    const float* __restrict__ h1, const float* __restrict__ V2,
    const float* __restrict__ c2, float* __restrict__ h2)
{
    __shared__ float x[256];
    int head = blockIdx.x >> 6;
    int g    = blockIdx.x & 63;
    int n    = threadIdx.x;
    x[n] = h1[((size_t)head * GG + g) * HH + n];
    __syncthreads();
    float acc = c2[n];
#pragma unroll 4
    for (int k = 0; k < 256; ++k) acc = fmaf(x[k], V2[(size_t)k * HH + n], acc);
    h2[((size_t)head * GG + g) * HH + n] = silu(acc);
}

__global__ __launch_bounds__(256) void head_out(
    const float* __restrict__ h2, const float* __restrict__ V3,
    const float* __restrict__ c3, float* __restrict__ out)
{
    __shared__ float partial[192];
    int t = threadIdx.x;
    if (t < 192) {
        int head = t >> 6; int g = t & 63;
        const float* row = h2 + ((size_t)head * GG + g) * HH;
        float acc = c3[0];
#pragma unroll 4
        for (int k = 0; k < 256; ++k) acc = fmaf(row[k], V3[k], acc);  // T == 1
        partial[t] = acc;
    }
    __syncthreads();
    if (t < 64) out[t] = partial[t] - partial[64 + t] - partial[128 + t];
}

// ---------------------------------------------------------------------------
extern "C" void kernel_launch(void* const* d_in, const int* in_sizes, int n_in,
                              void* d_out, int out_size, void* d_ws, size_t ws_size,
                              hipStream_t stream)
{
    const float* nf0 = (const float*)d_in[0];
    const float* ef0 = (const float*)d_in[1];
    const int*   ei0 = (const int*)  d_in[2];
    const int*   bt0 = (const int*)  d_in[3];
    const float* nf1 = (const float*)d_in[4];
    const float* ef1 = (const float*)d_in[5];
    const int*   ei1 = (const int*)  d_in[6];
    const int*   bt1 = (const int*)  d_in[7];
    const float* nf2 = (const float*)d_in[8];
    const float* ef2 = (const float*)d_in[9];
    const int*   ei2 = (const int*)  d_in[10];
    const int*   bt2 = (const int*)  d_in[11];
    const float* W1  = (const float*)d_in[12];
    const float* b1  = (const float*)d_in[13];
    const float* W2  = (const float*)d_in[14];
    const float* b2  = (const float*)d_in[15];
    const float* W3  = (const float*)d_in[16];
    const float* b3  = (const float*)d_in[17];
    const float* V1  = (const float*)d_in[18];
    const float* c1  = (const float*)d_in[19];
    const float* V2  = (const float*)d_in[20];
    const float* c2  = (const float*)d_in[21];
    const float* V3  = (const float*)d_in[22];
    const float* c3  = (const float*)d_in[23];

    // Workspace layout (≈1.2 MB total):
    uint8_t*  ws     = (uint8_t*)d_ws;
    uint32_t* Wp     = (uint32_t*)(ws);            // 3*32768 u32   = 384 KB
    float*    pooled = (float*)(ws + 393216);      // 6*64*256 f32  = 384 KB
    float*    h1     = (float*)(ws + 786432);      // 3*64*256 f32  = 192 KB
    float*    h2     = (float*)(ws + 983040);      // 3*64*256 f32  = 192 KB

    pack_weights_zero<<<384, 256, 0, stream>>>(W1, W2, W3, Wp, pooled);

    const int nblocks = (TOTAL_ROWS + 63) / 64;    // 7969 blocks of 64 rows
    fused_mlp1_scatter<<<nblocks, 256, 0, stream>>>(
        nf0, ef0, ei0, bt0, nf1, ef1, ei1, bt1, nf2, ef2, ei2, bt2,
        Wp, b1, b2, b3, pooled);

    head_mlp2_l1<<<192, 256, 0, stream>>>(pooled, V1, c1, h1);
    head_mlp2_l2<<<192, 256, 0, stream>>>(h1, V2, c2, h2);
    head_out<<<1, 256, 0, stream>>>(h2, V3, c3, (float*)d_out);
}
